// SelfAttention_3994319585291
// MI455X (gfx1250) — matrix operand order repaired and checked by reading, not compiled, not run on hardware
//
#include <hip/hip_runtime.h>
#include <hip/hip_bf16.h>
#include <math.h>

// ---------------------------------------------------------------------------
// MLA self-attention for MI455X (gfx1250)
//   - bf16 WMMA (16x16x32, f32 accum) for all 6 projection GEMMs + attention
//   - bf16 intermediates in global memory (halves HBM traffic)
//   - global_load_async_to_lds_b128 (ASYNCcnt) for bf16 tile staging
// ---------------------------------------------------------------------------

typedef __attribute__((ext_vector_type(16))) __bf16 v16bf;
typedef __attribute__((ext_vector_type(8)))  float  v8f;

union Frag16 { v16bf v; uint4 q[2]; };

__device__ inline v16bf ld_frag(const __bf16* p0, const __bf16* p1) {
    Frag16 f;
    f.q[0] = *reinterpret_cast<const uint4*>(p0);   // 16B, 16B-aligned
    f.q[1] = *reinterpret_cast<const uint4*>(p1);
    return f.v;
}

__device__ inline void cvt4(__bf16* dst, float4 v) {
    dst[0] = (__bf16)v.x; dst[1] = (__bf16)v.y;
    dst[2] = (__bf16)v.z; dst[3] = (__bf16)v.w;
}

// 16-byte async copy global -> LDS (per-lane), tracked by ASYNCcnt.
__device__ inline void async_cp16(const void* lds, const void* gptr) {
    const unsigned ldsOff = (unsigned)(unsigned long long)lds;  // low 32b = LDS offset
    asm volatile("global_load_async_to_lds_b128 %0, %1, off"
                 :: "v"(ldsOff), "v"(gptr) : "memory");
}
__device__ inline void async_wait0() {
    asm volatile("s_wait_asynccnt 0" ::: "memory");
}

__device__ inline void store_out(float*  C, size_t i, float v) { C[i] = v; }
__device__ inline void store_out(__bf16* C, size_t i, float v) { C[i] = (__bf16)v; }

#define KC 32

// C[M x N] = X[M x K] * W[N x K]^T + bias[N]
// TX = float (cvt staging) or __bf16 (async staging); TO = float or __bf16.
// transposedOut: C[n*ldOut + m]  (used for Vt). M%64==0, K%32==0; N guarded.
template <typename TX, typename TO>
__global__ void __launch_bounds__(256)
gemm_wmma(const TX* __restrict__ X, const float* __restrict__ W,
          const float* __restrict__ bias, TO* __restrict__ C,
          int M, int N, int K, int transposedOut, int ldOut)
{
    __shared__ __align__(16) __bf16 Xs[64][KC];    //  4 KB
    __shared__ __align__(16) __bf16 Ws[256][KC];   // 16 KB

    const int tid  = threadIdx.x;
    const int lane = tid & 31;
    const int wave = tid >> 5;
    const int wm   = wave >> 2;          // 0..1  (M direction, 32 rows each)
    const int wn   = wave & 3;           // 0..3  (N direction, 64 cols each)
    const int hi   = (lane >= 16) ? 1 : 0;
    const int lan  = lane & 15;

    const int m0 = blockIdx.y * 64;
    const int n0 = blockIdx.x * 256;

    v8f acc[2][4];
    for (int i = 0; i < 2; ++i)
        for (int j = 0; j < 4; ++j)
            acc[i][j] = (v8f){0.f,0.f,0.f,0.f,0.f,0.f,0.f,0.f};

    // staging maps
    const int xr = tid >> 2, xc = (tid & 3) * 8;   // X tile: 64 rows, 8 elems/thread
    const int wr = tid,      wc = 0;               // W tile: 256 rows, 32 f32/thread
    const int wn_g = n0 + wr;

    for (int k0 = 0; k0 < K; k0 += KC) {
        if constexpr (sizeof(TX) == 2) {           // bf16 X: async 16B copy
            const TX* src = X + (size_t)(m0 + xr) * K + k0 + xc;
            async_cp16(&Xs[xr][xc], src);
            if (k0 + KC < K) __builtin_prefetch(src + KC, 0, 1);
        } else {                                   // f32 X: load + cvt + ds store
            const float* src = (const float*)X + (size_t)(m0 + xr) * K + k0 + xc;
            float4 a = *reinterpret_cast<const float4*>(src);
            float4 b = *reinterpret_cast<const float4*>(src + 4);
            cvt4(&Xs[xr][xc],     a);
            cvt4(&Xs[xr][xc + 4], b);
            if (k0 + KC < K) __builtin_prefetch(src + KC, 0, 1);
        }
        {   // W tile (f32 weights -> bf16 LDS), one row per thread
            __bf16* dst = &Ws[wr][wc];
            if (wn_g < N) {
                const float* src = W + (size_t)wn_g * K + k0;
                for (int p = 0; p < 8; ++p) {
                    float4 a = *reinterpret_cast<const float4*>(src + p * 4);
                    cvt4(dst + p * 4, a);
                }
                if (k0 + KC < K) __builtin_prefetch(src + KC, 0, 1);
            } else {
                uint4 z = {0u,0u,0u,0u};
                uint4* dz = reinterpret_cast<uint4*>(dst);
                dz[0] = z; dz[1] = z; dz[2] = z; dz[3] = z;
            }
        }
        if constexpr (sizeof(TX) == 2) async_wait0();
        __syncthreads();

        v16bf afr[2], bfr[4];
        for (int t = 0; t < 2; ++t) {
            const int m = wm * 32 + t * 16 + lan;            // A rows: m = lane&15
            afr[t] = ld_frag(&Xs[m][hi ? 8 : 0], &Xs[m][16 + (hi ? 8 : 0)]);
        }
        for (int t = 0; t < 4; ++t) {
            const int n = wn * 64 + t * 16 + lan;            // B cols: n = lane&15
            bfr[t] = ld_frag(&Ws[n][hi ? 8 : 0], &Ws[n][16 + (hi ? 8 : 0)]);
        }
        for (int i = 0; i < 2; ++i)
            for (int j = 0; j < 4; ++j)
                acc[i][j] = __builtin_amdgcn_wmma_f32_16x16x32_bf16(
                    false, afr[i], false, bfr[j], (short)0, acc[i][j], false, false);
        __syncthreads();
    }

    // epilogue: bias + store (C layout: n = lane&15, m = r + hi*8)
    for (int j = 0; j < 4; ++j) {
        const int n_g = n0 + wn * 64 + j * 16 + lan;
        if (n_g >= N) continue;
        const float bv = bias[n_g];
        for (int i = 0; i < 2; ++i) {
            for (int r = 0; r < 8; ++r) {
                const int m_g = m0 + wm * 32 + i * 16 + r + (hi ? 8 : 0);
                const float val = acc[i][j][r] + bv;
                if (transposedOut) store_out(C, (size_t)n_g * ldOut + m_g, val);
                else               store_out(C, (size_t)m_g * ldOut + n_g, val);
            }
        }
    }
}

// RoPE in place on bf16 [S, H*HD]; scale folds 1/sqrt(HD) for Q.
__global__ void rope_kernel(__bf16* __restrict__ t, const float* __restrict__ cosb,
                            const float* __restrict__ sinb, float scale)
{
    const int idx = blockIdx.x * blockDim.x + threadIdx.x;  // S*H*64 threads
    const int d = idx & 63;
    const int h = (idx >> 6) & 15;
    const int s = idx >> 10;
    __bf16* p = t + (size_t)s * 2048 + h * 128;
    const float x1 = (float)p[d], x2 = (float)p[d + 64];
    const float c1 = cosb[s * 128 + d],      sn1 = sinb[s * 128 + d];
    const float c2 = cosb[s * 128 + d + 64], sn2 = sinb[s * 128 + d + 64];
    p[d]      = (__bf16)((x1 * c1 - x2 * sn1) * scale);
    p[d + 64] = (__bf16)((x2 * c2 + x1 * sn2) * scale);
}

// Flash attention: one block = (head h, 128 query rows), 8 waves x 16 rows.
// q,k: bf16 [S, H*HD] (q pre-scaled); vt: bf16 [H*HD, S]; attn: bf16 [S, H*HD]
__global__ void __launch_bounds__(256)
flash_attn(const __bf16* __restrict__ q, const __bf16* __restrict__ k,
           const __bf16* __restrict__ vt, __bf16* __restrict__ attn)
{
    __shared__ __align__(16) __bf16 Qs[128][128];  // 32 KB
    __shared__ __align__(16) __bf16 Ks[32][128];   //  8 KB
    __shared__ __align__(16) __bf16 Vs[128][32];   //  8 KB (rows = d, cols = kv)
    __shared__ __align__(16) __bf16 Ps[8][16][32]; //  8 KB per-wave P scratch

    const int tid  = threadIdx.x;
    const int lane = tid & 31;
    const int wave = tid >> 5;
    const int hi   = (lane >= 16) ? 1 : 0;
    const int lan  = lane & 15;
    const int h    = blockIdx.y;
    const int q0   = blockIdx.x * 128;

    // stage Q tile (128x128 bf16 = 2048 x 16B) via async copies
    for (int it = 0; it < 8; ++it) {
        const int idx = tid + it * 256;            // 0..2047
        const int row = idx >> 4, c8 = (idx & 15) * 8;
        async_cp16(&Qs[row][c8], q + (size_t)(q0 + row) * 2048 + h * 128 + c8);
    }
    async_wait0();
    __syncthreads();

    // Q A-fragments held in registers for the whole kernel (4 K-chunks of 32)
    v16bf qa[4];
    {
        const int m = wave * 16 + lan;
        for (int c = 0; c < 4; ++c)
            qa[c] = ld_frag(&Qs[m][c * 32 + (hi ? 8 : 0)],
                            &Qs[m][c * 32 + 16 + (hi ? 8 : 0)]);
    }

    v8f o[8];
    for (int t = 0; t < 8; ++t) o[t] = (v8f){0.f,0.f,0.f,0.f,0.f,0.f,0.f,0.f};
    float mstat[8], lstat[8];
    for (int r = 0; r < 8; ++r) { mstat[r] = -1e30f; lstat[r] = 0.f; }

    for (int kv0 = 0; kv0 < 2048; kv0 += 32) {
        __syncthreads();   // previous iteration's Ks/Vs reads are done
        // K tile 32x128 (512 x 16B) + Vt tile 128x32 (512 x 16B), async
        for (int it = 0; it < 2; ++it) {
            const int idx = tid + it * 256;        // 0..511
            const int kr = idx >> 4, kc = (idx & 15) * 8;
            async_cp16(&Ks[kr][kc], k + (size_t)(kv0 + kr) * 2048 + h * 128 + kc);
            const int vr = idx >> 2, vc = (idx & 3) * 8;
            async_cp16(&Vs[vr][vc], vt + (size_t)(h * 128 + vr) * 2048 + kv0 + vc);
        }
        async_wait0();
        __syncthreads();

        // S = Q * K^T  (16 q-rows x 32 kv-cols per wave)
        v8f st[2];
        st[0] = (v8f){0.f,0.f,0.f,0.f,0.f,0.f,0.f,0.f};
        st[1] = (v8f){0.f,0.f,0.f,0.f,0.f,0.f,0.f,0.f};
        for (int c = 0; c < 4; ++c) {
            for (int j = 0; j < 2; ++j) {
                const int n = j * 16 + lan;
                v16bf kb = ld_frag(&Ks[n][c * 32 + (hi ? 8 : 0)],
                                   &Ks[n][c * 32 + 16 + (hi ? 8 : 0)]);
                st[j] = __builtin_amdgcn_wmma_f32_16x16x32_bf16(
                    false, qa[c], false, kb, (short)0, st[j], false, false);
            }
        }

        // online softmax (row m = r + hi*8 lives in one 16-lane half)
        float pr[2][8];
        for (int r = 0; r < 8; ++r) {
            float mx = fmaxf(st[0][r], st[1][r]);
            for (int off = 8; off >= 1; off >>= 1)
                mx = fmaxf(mx, __shfl_xor(mx, off, 32));
            const float mnew  = fmaxf(mstat[r], mx);
            const float alpha = __expf(mstat[r] - mnew);
            float sum = 0.f;
            for (int j = 0; j < 2; ++j) {
                const float p = __expf(st[j][r] - mnew);
                pr[j][r] = p; sum += p;
            }
            for (int off = 8; off >= 1; off >>= 1)
                sum += __shfl_xor(sum, off, 32);
            lstat[r] = lstat[r] * alpha + sum;
            mstat[r] = mnew;
            for (int t = 0; t < 8; ++t) o[t][r] *= alpha;
        }

        // P: C-layout -> A-layout via per-wave LDS round trip
        for (int r = 0; r < 8; ++r) {
            const int m = r + (hi ? 8 : 0);
            Ps[wave][m][lan]      = (__bf16)pr[0][r];
            Ps[wave][m][16 + lan] = (__bf16)pr[1][r];
        }
        asm volatile("s_wait_dscnt 0" ::: "memory");
        v16bf pa = ld_frag(&Ps[wave][lan][hi ? 8 : 0],
                           &Ps[wave][lan][16 + (hi ? 8 : 0)]);

        // O += P * V  (8 d-tiles of 16)
        for (int t = 0; t < 8; ++t) {
            const int n = t * 16 + lan;            // d index within head
            v16bf vb = ld_frag(&Vs[n][hi ? 8 : 0], &Vs[n][16 + (hi ? 8 : 0)]);
            o[t] = __builtin_amdgcn_wmma_f32_16x16x32_bf16(
                false, pa, false, vb, (short)0, o[t], false, false);
        }
    }

    // normalize + store bf16
    for (int t = 0; t < 8; ++t) {
        const int n_g = h * 128 + t * 16 + lan;
        for (int r = 0; r < 8; ++r) {
            const int m_g = q0 + wave * 16 + r + (hi ? 8 : 0);
            attn[(size_t)m_g * 2048 + n_g] = (__bf16)(o[t][r] / lstat[r]);
        }
    }
}

// ---------------------------------------------------------------------------

extern "C" void kernel_launch(void* const* d_in, const int* in_sizes, int n_in,
                              void* d_out, int out_size, void* d_ws, size_t ws_size,
                              hipStream_t stream)
{
    const float* x        = (const float*)d_in[0];
    const float* rope_cos = (const float*)d_in[1];
    const float* rope_sin = (const float*)d_in[2];
    const float* wq_down  = (const float*)d_in[3];
    const float* bq_down  = (const float*)d_in[4];
    const float* wq_up    = (const float*)d_in[5];
    const float* bq_up    = (const float*)d_in[6];
    const float* wkv_down = (const float*)d_in[7];
    const float* bkv_down = (const float*)d_in[8];
    const float* wk_up    = (const float*)d_in[9];
    const float* bk_up    = (const float*)d_in[10];
    const float* wv_up    = (const float*)d_in[11];
    const float* bv_up    = (const float*)d_in[12];
    const float* wo       = (const float*)d_in[13];
    const float* bo       = (const float*)d_in[14];
    float* out = (float*)d_out;

    // workspace layout: ~38 MB (c_q/c_kv f32, q/k/vt/attn bf16)
    char* wsb = (char*)d_ws;
    float*  c_q  = (float*)wsb;   wsb += (size_t)2048 * 96   * 4;
    float*  c_kv = (float*)wsb;   wsb += (size_t)2048 * 512  * 4;
    __bf16* qb   = (__bf16*)wsb;  wsb += (size_t)2048 * 2048 * 2;
    __bf16* kb   = (__bf16*)wsb;  wsb += (size_t)2048 * 2048 * 2;
    __bf16* vtb  = (__bf16*)wsb;  wsb += (size_t)2048 * 2048 * 2;  // [H*HD][S]
    __bf16* attn = (__bf16*)wsb;

    const dim3 blk(256);
    // c_q  = x @ wq_down^T + b           (M=2048, N=96,  K=2048)  f32 out
    gemm_wmma<float, float><<<dim3(1, 32), blk, 0, stream>>>(
        x, wq_down, bq_down, c_q, 2048, 96, 2048, 0, 96);
    // c_kv = x @ wkv_down^T + b          (N=512)                  f32 out
    gemm_wmma<float, float><<<dim3(2, 32), blk, 0, stream>>>(
        x, wkv_down, bkv_down, c_kv, 2048, 512, 2048, 0, 512);
    // q = c_q @ wq_up^T + b              (K=96)                   bf16 out
    gemm_wmma<float, __bf16><<<dim3(8, 32), blk, 0, stream>>>(
        c_q, wq_up, bq_up, qb, 2048, 2048, 96, 0, 2048);
    // k = c_kv @ wk_up^T + b             (K=512)                  bf16 out
    gemm_wmma<float, __bf16><<<dim3(8, 32), blk, 0, stream>>>(
        c_kv, wk_up, bk_up, kb, 2048, 2048, 512, 0, 2048);
    // v^T = (c_kv @ wv_up^T + b)^T       stored [H*HD][S]         bf16 out
    gemm_wmma<float, __bf16><<<dim3(8, 32), blk, 0, stream>>>(
        c_kv, wv_up, bv_up, vtb, 2048, 2048, 512, 1, 2048);

    // RoPE (q also scaled by 1/sqrt(128))
    const int rthreads = 2048 * 16 * 64;
    rope_kernel<<<rthreads / 256, 256, 0, stream>>>(qb, rope_cos, rope_sin,
                                                    0.08838834764831845f);
    rope_kernel<<<rthreads / 256, 256, 0, stream>>>(kb, rope_cos, rope_sin, 1.0f);

    // fused softmax attention (async-LDS staged, WMMA QK^T and PV)
    flash_attn<<<dim3(16, 16), blk, 0, stream>>>(qb, kb, vtb, attn);

    // out = attn @ wo^T + bo             (bf16 X via async staging, f32 out)
    gemm_wmma<__bf16, float><<<dim3(8, 32), blk, 0, stream>>>(
        attn, wo, bo, out, 2048, 2048, 2048, 0, 2048);
}
